// HybridQuantumKernelNet_65481071398094
// MI455X (gfx1250) — matrix operand order repaired and checked by reading, not compiled, not run on hardware
//
#include <hip/hip_runtime.h>
#include <hip/hip_bf16.h>

// ---------------------------------------------------------------------------
// HybridQuantumKernelNet for MI455X (gfx1250, wave32, WMMA)
//
// Pipeline:
//   k_conv1     : x[64,3,250,250] -> relu(conv 5x5 s2 p1) -> c1[64,6,124,124]
//   k_conv2     : (maxpool2 s1 fused on read of c1) -> relu(conv 3x3 s2 p1)
//                 -> c2[64,15,62,62]
//   k_pool2h16  : maxpool2 s1 on c2, flatten, cast f16, zero-pad K to 56320
//                 -> h16[64, KPAD]
//   k_w16       : fc1_w f32[120,55815] -> f16[128, KPAD] zero-padded
//   k_fc1wmma   : split-K (16 slices x exactly 110 k-steps, compile-time
//                 constant trip count for unroll+pipelining) via
//                 v_wmma_f32_16x16x32_f16 -> partials[16,64,128]
//   k_fc1reduce : sum 16 partials + bias + relu -> fc1o[64,120]
//   k_fc2       : dense 120->84 + relu
//   k_head      : fc3 (84->1) + cos((z-p)/2)^4 kernel + 10->1 + sigmoid
// ---------------------------------------------------------------------------

typedef __attribute__((ext_vector_type(16))) _Float16 v16h;
typedef __attribute__((ext_vector_type(8)))  _Float16 v8h;
typedef __attribute__((ext_vector_type(8)))  float    v8f;

#define KREAL 55815      // 15*61*61
#define KPAD  56320      // 1760 k-steps of 32; 16B-aligned rows
#define NPAD  128        // 120 padded to 8 N-tiles
#define KSTEPS 1760      // KPAD / 32
#define NSLICES 16       // split-K factor -> 512 independent waves
#define STEPS_PER_SLICE 110   // KSTEPS / NSLICES, exact -> constant trip count

#define C1_ELEMS (64*6*124*124)     // 5,904,384
#define C2_ELEMS (64*15*62*62)      // 3,690,240
#define H16_ELEMS (64*KPAD)         // 3,604,480
#define W16_ELEMS (NPAD*KPAD)       // 7,208,960

// ---------------------------- conv1 + relu ---------------------------------
__global__ void k_conv1(const float* __restrict__ x,
                        const float* __restrict__ w,
                        const float* __restrict__ bias,
                        float* __restrict__ out) {
    int idx = blockIdx.x * blockDim.x + threadIdx.x;
    if (idx >= C1_ELEMS) return;
    int ow = idx % 124; int t = idx / 124;
    int oh = t % 124;   t /= 124;
    int c  = t % 6;     int b = t / 6;

    const float* xb = x + (size_t)b * 3 * 250 * 250;
    const float* wc = w + c * 3 * 5 * 5;
    float acc = bias[c];
    #pragma unroll
    for (int ci = 0; ci < 3; ++ci) {
        #pragma unroll
        for (int kh = 0; kh < 5; ++kh) {
            int ih = 2 * oh - 1 + kh;
            if (ih < 0 || ih >= 250) continue;
            #pragma unroll
            for (int kw = 0; kw < 5; ++kw) {
                int iw = 2 * ow - 1 + kw;
                if (iw < 0 || iw >= 250) continue;
                acc += xb[((size_t)ci * 250 + ih) * 250 + iw] *
                       wc[(ci * 5 + kh) * 5 + kw];
            }
        }
    }
    out[idx] = acc > 0.f ? acc : 0.f;
}

// ---------------- conv2 + relu (maxpool2 s1 fused on input) ----------------
__device__ __forceinline__ float pool1_at(const float* __restrict__ c1,
                                          int b, int ci, int ih, int iw) {
    // pool1[ih,iw] = max 2x2 over c1[124x124]; valid for ih,iw in [0,123)
    const float* p = c1 + (((size_t)b * 6 + ci) * 124 + ih) * 124 + iw;
    float m0 = fmaxf(p[0], p[1]);
    float m1 = fmaxf(p[124], p[125]);
    return fmaxf(m0, m1);
}

__global__ void k_conv2(const float* __restrict__ c1,
                        const float* __restrict__ w,
                        const float* __restrict__ bias,
                        float* __restrict__ out) {
    int idx = blockIdx.x * blockDim.x + threadIdx.x;
    if (idx >= C2_ELEMS) return;
    int ow = idx % 62; int t = idx / 62;
    int oh = t % 62;   t /= 62;
    int c  = t % 15;   int b = t / 15;

    float acc = bias[c];
    #pragma unroll
    for (int ci = 0; ci < 6; ++ci) {
        #pragma unroll
        for (int kh = 0; kh < 3; ++kh) {
            int ih = 2 * oh - 1 + kh;
            if (ih < 0 || ih >= 123) continue;
            #pragma unroll
            for (int kw = 0; kw < 3; ++kw) {
                int iw = 2 * ow - 1 + kw;
                if (iw < 0 || iw >= 123) continue;
                acc += pool1_at(c1, b, ci, ih, iw) *
                       w[((c * 6 + ci) * 3 + kh) * 3 + kw];
            }
        }
    }
    out[idx] = acc > 0.f ? acc : 0.f;
}

// ------------- maxpool2 s1 + flatten + f16 cast + K zero-pad ---------------
__global__ void k_pool2h16(const float* __restrict__ c2,
                           _Float16* __restrict__ h16) {
    int idx = blockIdx.x * blockDim.x + threadIdx.x;
    if (idx >= H16_ELEMS) return;
    int kp = idx % KPAD;
    int b  = idx / KPAD;
    _Float16 v = (_Float16)0.f;
    if (kp < KREAL) {
        int c = kp / 3721;            // 61*61
        int r = kp % 3721;
        int y = r / 61;
        int x = r % 61;
        const float* p = c2 + (((size_t)b * 15 + c) * 62 + y) * 62 + x;
        float m = fmaxf(fmaxf(p[0], p[1]), fmaxf(p[62], p[63]));
        v = (_Float16)m;
    }
    h16[idx] = v;
}

// --------------------- fc1 weight f32 -> f16 with padding ------------------
__global__ void k_w16(const float* __restrict__ w,
                      _Float16* __restrict__ w16) {
    int idx = blockIdx.x * blockDim.x + threadIdx.x;
    if (idx >= W16_ELEMS) return;
    int kp = idx % KPAD;
    int n  = idx / KPAD;
    _Float16 v = (_Float16)0.f;
    if (n < 120 && kp < KREAL) v = (_Float16)w[(size_t)n * KREAL + kp];
    w16[idx] = v;
}

// ------------- fc1 GEMM, split-K: v_wmma_f32_16x16x32_f16 ------------------
// 512 blocks x 32 threads (1 wave each): wave = (K-slice, 16x16 MxN tile).
// A = h16[64, KPAD] row-major (f16). B = w16[NPAD, KPAD]: B[k][n] = w16[n][k],
// i.e. a lane's B column slice is contiguous in w16's row n.
// Constant 110-step trip count -> compiler unrolls & software-pipelines the
// load/WMMA stream (s_wait_loadcnt > 0 between v_wmma ops).
__global__ void k_fc1wmma(const _Float16* __restrict__ h16,
                          const _Float16* __restrict__ w16,
                          float* __restrict__ partial) {
    const int wid   = blockIdx.x;     // 0..511
    const int tile  = wid & 31;       // 0..31
    const int slice = wid >> 5;       // 0..15
    const int mt = tile >> 3;         // 0..3  (M tile)
    const int nt = tile & 7;          // 0..7  (N tile)
    const int lane = threadIdx.x;     // 0..31
    const int half = lane >> 4;       // 0 | 1
    const int l15  = lane & 15;

    const size_t kbase = (size_t)slice * STEPS_PER_SLICE * 32;
    // A (16x32 f16) per-lane addressing (ISA 7.12.2):
    //   lanes 0-15 : M=l15, chunks K=[k0..k0+7] and [k0+16..k0+23]
    //   lanes 16-31: M=l15, chunks K=[k0+8..k0+15] and [k0+24..k0+31]
    const _Float16* aRow = h16 + (size_t)(mt * 16 + l15) * KPAD + half * 8 + kbase;
    // B (32x16 f16): column N=l15, same K chunks as A (lanes 0-15: K=0..7,16..23; lanes 16-31: K=8..15,24..31)
    const _Float16* bRow = w16 + (size_t)(nt * 16 + l15) * KPAD + half * 8 + kbase;

    v8f c = {};
    #pragma unroll 5
    for (int s = 0; s < STEPS_PER_SLICE; ++s) {
        const int k0 = s * 32;
        v8h a0 = *(const v8h*)(aRow + k0);
        v8h a1 = *(const v8h*)(aRow + k0 + 16);
        v8h b0 = *(const v8h*)(bRow + k0);
        v8h b1 = *(const v8h*)(bRow + k0 + 16);
        __builtin_prefetch(bRow + k0 + 256, 0, 3);   // global_prefetch_b8
        __builtin_prefetch(aRow + k0 + 256, 0, 3);
        v16h a = __builtin_shufflevector(a0, a1, 0,1,2,3,4,5,6,7,8,9,10,11,12,13,14,15);
        v16h b = __builtin_shufflevector(b0, b1, 0,1,2,3,4,5,6,7,8,9,10,11,12,13,14,15);
        // (neg_a, A, neg_b, B, c_mod, C, reuse_a, reuse_b)
        c = __builtin_amdgcn_wmma_f32_16x16x32_f16(false, a, false, b,
                                                   (short)0, c, false, false);
    }

    // C/D layout: VGPR r, lanes 0-15 -> M=r, N=lane; lanes 16-31 -> M=8+r
    const int n = nt * 16 + l15;
    #pragma unroll
    for (int r = 0; r < 8; ++r) {
        int m = mt * 16 + r + half * 8;
        partial[((size_t)slice * 64 + m) * NPAD + n] = c[r];
    }
}

// ---------------- split-K reduction + bias + relu --------------------------
__global__ void k_fc1reduce(const float* __restrict__ partial,
                            const float* __restrict__ fc1_b,
                            float* __restrict__ out) {
    int idx = blockIdx.x * blockDim.x + threadIdx.x;
    if (idx >= 64 * 120) return;
    int n = idx % 120;
    int m = idx / 120;
    float acc = fc1_b[n];
    #pragma unroll
    for (int s = 0; s < NSLICES; ++s)
        acc += partial[((size_t)s * 64 + m) * NPAD + n];
    out[idx] = acc > 0.f ? acc : 0.f;
}

// ------------------------------- fc2 + relu --------------------------------
__global__ void k_fc2(const float* __restrict__ in,
                      const float* __restrict__ w,
                      const float* __restrict__ bias,
                      float* __restrict__ out) {
    int idx = blockIdx.x * blockDim.x + threadIdx.x;
    if (idx >= 64 * 84) return;
    int j = idx % 84;
    int b = idx / 84;
    float acc = bias[j];
    #pragma unroll 8
    for (int i = 0; i < 120; ++i)
        acc += in[b * 120 + i] * w[j * 120 + i];
    out[idx] = acc > 0.f ? acc : 0.f;
}

// -------------- fc3 + quantum kernel sim + head + sigmoid ------------------
__global__ void k_head(const float* __restrict__ h2,
                       const float* __restrict__ fc3_w,
                       const float* __restrict__ fc3_b,
                       const float* __restrict__ proto,
                       const float* __restrict__ kh_w,
                       const float* __restrict__ kh_b,
                       float* __restrict__ out) {
    int b = blockIdx.x * blockDim.x + threadIdx.x;
    if (b >= 64) return;
    float z = fc3_b[0];
    #pragma unroll 4
    for (int j = 0; j < 84; ++j) z += h2[b * 84 + j] * fc3_w[j];
    float logit = kh_b[0];
    #pragma unroll
    for (int p = 0; p < 10; ++p) {
        float cs = cosf((z - proto[p]) * 0.5f);
        float c2 = cs * cs;
        logit += c2 * c2 * kh_w[p];
    }
    float prob = 1.f / (1.f + expf(-logit));
    out[b * 2 + 0] = prob;
    out[b * 2 + 1] = 1.f - prob;
}

// ------------------------------ launcher -----------------------------------
extern "C" void kernel_launch(void* const* d_in, const int* in_sizes, int n_in,
                              void* d_out, int out_size, void* d_ws, size_t ws_size,
                              hipStream_t stream) {
    const float* x       = (const float*)d_in[0];
    const float* conv1_w = (const float*)d_in[1];
    const float* conv1_b = (const float*)d_in[2];
    const float* conv2_w = (const float*)d_in[3];
    const float* conv2_b = (const float*)d_in[4];
    const float* fc1_w   = (const float*)d_in[5];
    const float* fc1_b   = (const float*)d_in[6];
    const float* fc2_w   = (const float*)d_in[7];
    const float* fc2_b   = (const float*)d_in[8];
    const float* fc3_w   = (const float*)d_in[9];
    const float* fc3_b   = (const float*)d_in[10];
    const float* proto   = (const float*)d_in[11];
    const float* kh_w    = (const float*)d_in[12];
    const float* kh_b    = (const float*)d_in[13];

    char* ws = (char*)d_ws;
    // 16B-aligned workspace carve-up (~61 MB total)
    float*    c1   = (float*)   (ws + 0);                       // 23,617,536 B
    float*    c2   = (float*)   (ws + 23617536);                // 14,760,960 B
    _Float16* h16  = (_Float16*)(ws + 38378496);                //  7,208,960 B
    _Float16* w16  = (_Float16*)(ws + 45587456);                // 14,417,920 B
    float*    fc1o = (float*)   (ws + 60005376);                //     30,720 B
    float*    fc2o = (float*)   (ws + 60036096);                //     21,504 B
    float*    part = (float*)   (ws + 60057600);                //    524,288 B

    const int T = 256;
    hipLaunchKernelGGL(k_conv1,    dim3((C1_ELEMS  + T - 1) / T), dim3(T), 0, stream,
                       x, conv1_w, conv1_b, c1);
    hipLaunchKernelGGL(k_w16,      dim3((W16_ELEMS + T - 1) / T), dim3(T), 0, stream,
                       fc1_w, w16);
    hipLaunchKernelGGL(k_conv2,    dim3((C2_ELEMS  + T - 1) / T), dim3(T), 0, stream,
                       c1, conv2_w, conv2_b, c2);
    hipLaunchKernelGGL(k_pool2h16, dim3((H16_ELEMS + T - 1) / T), dim3(T), 0, stream,
                       c2, h16);
    hipLaunchKernelGGL(k_fc1wmma,  dim3(32 * NSLICES), dim3(32), 0, stream,
                       h16, w16, part);
    hipLaunchKernelGGL(k_fc1reduce,dim3((64 * 120 + T - 1) / T), dim3(T), 0, stream,
                       part, fc1_b, fc1o);
    hipLaunchKernelGGL(k_fc2,      dim3((64 * 84 + T - 1) / T), dim3(T), 0, stream,
                       fc1o, fc2_w, fc2_b, fc2o);
    hipLaunchKernelGGL(k_head,     dim3(1), dim3(64), 0, stream,
                       fc2o, fc3_w, fc3_b, proto, kh_w, kh_b, (float*)d_out);
}
